// LinkPredictionModel_46145128628314
// MI455X (gfx1250) — matrix operand order repaired and checked by reading, not compiled, not run on hardware
//
#include <hip/hip_runtime.h>
#include <hip/hip_bf16.h>

typedef __attribute__((ext_vector_type(16))) _Float16 v16h;
typedef __attribute__((ext_vector_type(8)))  _Float16 v8h;
typedef __attribute__((ext_vector_type(8)))  float    v8f;

#define N_NODES  50000
#define N_EDGES  800000
#define D_IN     64
#define HID      128
#define OUT_D    64
#define N_LABEL  100000
#define NPAD     50016        // N_NODES + 16, a multiple of 16
#define A1_COLS  192          // [x | agg_f | agg_r]
#define A2_COLS  384          // [h | aggh_f | aggh_r]
#define COMB_COLS 192         // [zs | zd | zs*zd]

// ---------------------------------------------------------------------------
// WMMA fragment loaders (ISA 7.12.2 layouts, wave32, 16-bit operands)
// A: 16x32 (MxK), row-major source with leading dim lda (halves).
// Per lane the 16 halves are two contiguous 16-byte runs: k = 0..7 and 16..23
// relative to (row = lane&15, base = (lane>>4)*8)  ->  two b128 loads.
__device__ inline v16h load_a_frag(const _Float16* __restrict__ A, int lda, int lane) {
  int m  = lane & 15;
  int hi = lane >> 4;
  const _Float16* p = A + m * lda + hi * 8;
  v8h lo  = *(const v8h*)(p);
  v8h hi8 = *(const v8h*)(p + 16);
  return __builtin_shufflevector(lo, hi8, 0, 1, 2, 3, 4, 5, 6, 7,
                                          8, 9, 10, 11, 12, 13, 14, 15);
}

// B: pre-swizzled fragment-major weights. Fragment `frag`, lane `lane` owns 16
// contiguous halves (32B, aligned) -> single v16h load (two b128).
__device__ inline v16h load_b_swz(const _Float16* __restrict__ Wsw, int frag, int lane) {
  return *(const v16h*)(Wsw + (((frag << 5) + lane) << 4));
}

// Swizzle map used by the weight-pack kernels:
//   out[((kt*NTILES + nt)*32 + lane)*16 + j]  <=  W[k][c]
//   with k = kt*32 + (lane>>4)*8 + (j&7) + (j>>3)*16 ,  c = nt*16 + (lane&15)
__device__ inline void swz_coords(int idx, int ntiles, int& k, int& c) {
  int j    = idx & 15;
  int lane = (idx >> 4) & 31;
  int frag = idx >> 9;
  int nt   = frag % ntiles;
  int kt   = frag / ntiles;
  k = kt * 32 + ((lane >> 4) << 3) + (j & 7) + ((j >> 3) << 4);
  c = nt * 16 + (lane & 15);
}

// ---------------------------------------------------------------------------
// Weight packing: fold the 4 per-relation SAGE matmuls into one GEMM per layer
// and emit WMMA-fragment-swizzled f16.
// W1 logical [192][128]: rows 0-63 = c1f_Wr + c1r_Wr, 64-127 = c1f_Wl, 128-191 = c1r_Wl
__global__ void pack_w1_kernel(const float* __restrict__ Wl_f, const float* __restrict__ Wr_f,
                               const float* __restrict__ Wl_r, const float* __restrict__ Wr_r,
                               const float* __restrict__ bl_f, const float* __restrict__ bl_r,
                               _Float16* __restrict__ W1, float* __restrict__ bias1) {
  int idx = blockIdx.x * blockDim.x + threadIdx.x;
  if (idx < 192 * 128) {
    int k, c;
    swz_coords(idx, 8, k, c);
    float v;
    if (k < 64)        v = Wr_f[k * 128 + c] + Wr_r[k * 128 + c];
    else if (k < 128)  v = Wl_f[(k - 64) * 128 + c];
    else               v = Wl_r[(k - 128) * 128 + c];
    W1[idx] = (_Float16)v;
  }
  if (idx < 128) bias1[idx] = bl_f[idx] + bl_r[idx];
}

// W2 logical [384][64]: rows 0-127 = c2f_Wr + c2r_Wr, 128-255 = c2f_Wl, 256-383 = c2r_Wl
__global__ void pack_w2_kernel(const float* __restrict__ Wl_f, const float* __restrict__ Wr_f,
                               const float* __restrict__ Wl_r, const float* __restrict__ Wr_r,
                               const float* __restrict__ bl_f, const float* __restrict__ bl_r,
                               _Float16* __restrict__ W2, float* __restrict__ bias2) {
  int idx = blockIdx.x * blockDim.x + threadIdx.x;
  if (idx < 384 * 64) {
    int k, c;
    swz_coords(idx, 4, k, c);
    float v;
    if (k < 128)       v = Wr_f[k * 64 + c] + Wr_r[k * 64 + c];
    else if (k < 256)  v = Wl_f[(k - 128) * 64 + c];
    else               v = Wl_r[(k - 256) * 64 + c];
    W2[idx] = (_Float16)v;
  }
  if (idx < 64) bias2[idx] = bl_f[idx] + bl_r[idx];
}

// Decoder weights: dW1 [192][64] (4 n-tiles), dW2 [64][32] (2 n-tiles)
__global__ void pack_dec_kernel(const float* __restrict__ dW1, const float* __restrict__ dW2,
                                _Float16* __restrict__ W1h, _Float16* __restrict__ W2h) {
  int idx = blockIdx.x * blockDim.x + threadIdx.x;
  if (idx < 192 * 64) {
    int k, c;
    swz_coords(idx, 4, k, c);
    W1h[idx] = (_Float16)dW1[k * 64 + c];
  } else if (idx < 192 * 64 + 64 * 32) {
    int i2 = idx - 192 * 64;
    int k, c;
    swz_coords(i2, 2, k, c);
    W2h[i2] = (_Float16)dW2[k * 32 + c];
  }
}

// ---------------------------------------------------------------------------
// Degree counts for both mean-aggregation directions (same graph both layers).
__global__ void deg_kernel(const int* __restrict__ ei,
                           float* __restrict__ deg_f, float* __restrict__ deg_r) {
  int e = blockIdx.x * blockDim.x + threadIdx.x;
  if (e >= N_EDGES) return;
  int s = ei[e], d = ei[N_EDGES + e];
  atomicAdd(&deg_f[d], 1.0f);   // forward relation: aggregate into dst
  atomicAdd(&deg_r[s], 1.0f);   // reverse relation: aggregate into src
}

// Edge scatter-add of D-dim features, both directions in one pass (L2-resident).
template <int D>
__global__ void scatter_kernel(const int* __restrict__ ei, const float* __restrict__ feat,
                               float* __restrict__ sum_f, float* __restrict__ sum_r) {
  const int CH = D / 4;
  int t = blockIdx.x * blockDim.x + threadIdx.x;
  if (t >= N_EDGES * CH) return;
  int e = t / CH;
  int c = (t % CH) * 4;
  int s = ei[e], d = ei[N_EDGES + e];
  const float4 xs = *(const float4*)(feat + (long long)s * D + c);
  const float4 xd = *(const float4*)(feat + (long long)d * D + c);
  float* pf = sum_f + (long long)d * D + c;
  float* pr = sum_r + (long long)s * D + c;
  atomicAdd(pf + 0, xs.x); atomicAdd(pf + 1, xs.y);
  atomicAdd(pf + 2, xs.z); atomicAdd(pf + 3, xs.w);
  atomicAdd(pr + 0, xd.x); atomicAdd(pr + 1, xd.y);
  atomicAdd(pr + 2, xd.z); atomicAdd(pr + 3, xd.w);
}

// Pack [feat | sum_f/deg_f | sum_r/deg_r] into an f16 A matrix, zero-padded rows.
template <int D>
__global__ void pack_a_kernel(const float* __restrict__ feat,
                              const float* __restrict__ sum_f, const float* __restrict__ deg_f,
                              const float* __restrict__ sum_r, const float* __restrict__ deg_r,
                              _Float16* __restrict__ A) {
  const int COLS = 3 * D;
  int idx = blockIdx.x * blockDim.x + threadIdx.x;
  if (idx >= NPAD * COLS) return;
  int n = idx / COLS, c = idx % COLS;
  float v = 0.0f;
  if (n < N_NODES) {
    if (c < D)            v = feat[(long long)n * D + c];
    else if (c < 2 * D)   v = sum_f[(long long)n * D + (c - D)]     / fmaxf(deg_f[n], 1.0f);
    else                  v = sum_r[(long long)n * D + (c - 2 * D)] / fmaxf(deg_r[n], 1.0f);
  }
  A[idx] = (_Float16)v;
}

// ---------------------------------------------------------------------------
// GEMM1: h = ReLU(LayerNorm(A1 @ W1 + bias1))
// A1: [NPAD x 192] f16 ; W1 swizzled [6 kt][8 nt] ; per-wave 16x128 tile
__global__ __launch_bounds__(128)
void gemm1_ln_relu_kernel(const _Float16* __restrict__ A, const _Float16* __restrict__ W,
                          const float* __restrict__ bias,
                          const float* __restrict__ g, const float* __restrict__ b,
                          float* __restrict__ h) {
  __shared__ float tile[4][16][132];
  int wave = threadIdx.x >> 5;
  int lane = threadIdx.x & 31;
  int rt = blockIdx.x * 4 + wave;          // wave-uniform tile id -> EXEC all-1 for WMMA
  const int NT = NPAD / 16;                // 3126
  if (rt < NT) {
    int m0 = rt * 16;
    v8f acc[8] = {};
    for (int kt = 0; kt < 6; ++kt) {
      v16h a = load_a_frag(A + (long long)m0 * A1_COLS + kt * 32, A1_COLS, lane);
#pragma unroll
      for (int nt = 0; nt < 8; ++nt) {
        v16h bb = load_b_swz(W, kt * 8 + nt, lane);
        acc[nt] = __builtin_amdgcn_wmma_f32_16x16x32_f16(
            false, a, false, bb, (short)0, acc[nt], false, false);
      }
    }
    int n = lane & 15, hi = lane >> 4;
#pragma unroll
    for (int nt = 0; nt < 8; ++nt) {
      int col = nt * 16 + n;
#pragma unroll
      for (int i = 0; i < 8; ++i)
        tile[wave][i + hi * 8][col] = acc[nt][i] + bias[col];
    }
  }
  __syncthreads();
  int r = threadIdx.x;
  if (r < 64) {
    int w = r >> 4, row = r & 15;
    int gr = (blockIdx.x * 4 + w) * 16 + row;
    if (gr < N_NODES) {
      float m = 0.0f;
      for (int c = 0; c < 128; ++c) m += tile[w][row][c];
      m *= (1.0f / 128.0f);
      float var = 0.0f;
      for (int c = 0; c < 128; ++c) { float d = tile[w][row][c] - m; var += d * d; }
      var *= (1.0f / 128.0f);
      float inv = rsqrtf(var + 1e-5f);
      for (int c = 0; c < 128; ++c) {
        float val = (tile[w][row][c] - m) * inv * g[c] + b[c];
        h[(long long)gr * 128 + c] = fmaxf(val, 0.0f);
      }
    }
  }
}

// GEMM2: z = L2normalize(A2 @ W2 + bias2). A2: [NPAD x 384], W2 swizzled [12 kt][4 nt]
__global__ __launch_bounds__(128)
void gemm2_l2norm_kernel(const _Float16* __restrict__ A, const _Float16* __restrict__ W,
                         const float* __restrict__ bias, float* __restrict__ z) {
  __shared__ float tile[4][16][68];
  int wave = threadIdx.x >> 5;
  int lane = threadIdx.x & 31;
  int rt = blockIdx.x * 4 + wave;
  const int NT = NPAD / 16;
  if (rt < NT) {
    int m0 = rt * 16;
    v8f acc[4] = {};
    for (int kt = 0; kt < 12; ++kt) {
      v16h a = load_a_frag(A + (long long)m0 * A2_COLS + kt * 32, A2_COLS, lane);
#pragma unroll
      for (int nt = 0; nt < 4; ++nt) {
        v16h bb = load_b_swz(W, kt * 4 + nt, lane);
        acc[nt] = __builtin_amdgcn_wmma_f32_16x16x32_f16(
            false, a, false, bb, (short)0, acc[nt], false, false);
      }
    }
    int n = lane & 15, hi = lane >> 4;
#pragma unroll
    for (int nt = 0; nt < 4; ++nt) {
      int col = nt * 16 + n;
#pragma unroll
      for (int i = 0; i < 8; ++i)
        tile[wave][i + hi * 8][col] = acc[nt][i] + bias[col];
    }
  }
  __syncthreads();
  int r = threadIdx.x;
  if (r < 64) {
    int w = r >> 4, row = r & 15;
    int gr = (blockIdx.x * 4 + w) * 16 + row;
    if (gr < N_NODES) {
      float ss = 0.0f;
      for (int c = 0; c < 64; ++c) { float v = tile[w][row][c]; ss += v * v; }
      float inv = 1.0f / fmaxf(sqrtf(ss), 1e-12f);
      for (int c = 0; c < 64; ++c)
        z[(long long)gr * 64 + c] = tile[w][row][c] * inv;
    }
  }
}

// Build decoder input comb = [zs | zd | zs*zd] in f16 (100000 rows, 192 cols)
__global__ void pack_comb_kernel(const int* __restrict__ eli, const float* __restrict__ z,
                                 _Float16* __restrict__ comb) {
  int idx = blockIdx.x * blockDim.x + threadIdx.x;
  if (idx >= N_LABEL * COMB_COLS) return;
  int i = idx / COMB_COLS, c = idx % COMB_COLS;
  int s = eli[i], d = eli[N_LABEL + i];
  float v;
  if (c < 64)        v = z[(long long)s * 64 + c];
  else if (c < 128)  v = z[(long long)d * 64 + (c - 64)];
  else               v = z[(long long)s * 64 + (c - 128)] * z[(long long)d * 64 + (c - 128)];
  comb[idx] = (_Float16)v;
}

// Fully fused decoder MLP: 192 -> relu 64 -> relu 32 -> 1 per 16-row tile.
__global__ __launch_bounds__(128)
void decoder_kernel(const _Float16* __restrict__ comb,
                    const _Float16* __restrict__ W1, const float* __restrict__ b1,
                    const _Float16* __restrict__ W2, const float* __restrict__ b2,
                    const float* __restrict__ W3, const float* __restrict__ b3,
                    float* __restrict__ out) {
  __shared__ _Float16 h1[4][16][64];
  __shared__ float    h2[4][16][36];
  int wave = threadIdx.x >> 5;
  int lane = threadIdx.x & 31;
  int rt = blockIdx.x * 4 + wave;
  const int NT = N_LABEL / 16;             // 6250 (exact)
  if (rt < NT) {
    int m0 = rt * 16;
    v8f acc[4] = {};
    for (int kt = 0; kt < 6; ++kt) {
      v16h a = load_a_frag(comb + (long long)m0 * COMB_COLS + kt * 32, COMB_COLS, lane);
#pragma unroll
      for (int nt = 0; nt < 4; ++nt) {
        v16h bb = load_b_swz(W1, kt * 4 + nt, lane);
        acc[nt] = __builtin_amdgcn_wmma_f32_16x16x32_f16(
            false, a, false, bb, (short)0, acc[nt], false, false);
      }
    }
    int n = lane & 15, hi = lane >> 4;
#pragma unroll
    for (int nt = 0; nt < 4; ++nt)
#pragma unroll
      for (int i = 0; i < 8; ++i)
        h1[wave][i + hi * 8][nt * 16 + n] =
            (_Float16)fmaxf(acc[nt][i] + b1[nt * 16 + n], 0.0f);
  }
  __syncthreads();
  if (rt < NT) {
    v8f acc2[2] = {};
#pragma unroll
    for (int kt = 0; kt < 2; ++kt) {
      v16h a = load_a_frag(&h1[wave][0][0] + kt * 32, 64, lane);  // ds_load_b128 x2
#pragma unroll
      for (int nt = 0; nt < 2; ++nt) {
        v16h bb = load_b_swz(W2, kt * 2 + nt, lane);
        acc2[nt] = __builtin_amdgcn_wmma_f32_16x16x32_f16(
            false, a, false, bb, (short)0, acc2[nt], false, false);
      }
    }
    int n = lane & 15, hi = lane >> 4;
#pragma unroll
    for (int nt = 0; nt < 2; ++nt)
#pragma unroll
      for (int i = 0; i < 8; ++i)
        h2[wave][i + hi * 8][nt * 16 + n] = fmaxf(acc2[nt][i] + b2[nt * 16 + n], 0.0f);
  }
  __syncthreads();
  int r = threadIdx.x;
  if (r < 64) {
    int w = r >> 4, row = r & 15;
    int gr = (blockIdx.x * 4 + w) * 16 + row;
    if (gr < N_LABEL) {
      float s = b3[0];
      for (int c = 0; c < 32; ++c) s += h2[w][row][c] * W3[c];
      out[gr] = s;
    }
  }
}

// ---------------------------------------------------------------------------
extern "C" void kernel_launch(void* const* d_in, const int* in_sizes, int n_in,
                              void* d_out, int out_size, void* d_ws, size_t ws_size,
                              hipStream_t stream) {
  (void)in_sizes; (void)n_in; (void)out_size; (void)ws_size;
  const float* x      = (const float*)d_in[0];
  const int*   ei     = (const int*)d_in[1];
  const int*   eli    = (const int*)d_in[2];
  const float* c1f_Wl = (const float*)d_in[3];
  const float* c1f_bl = (const float*)d_in[4];
  const float* c1f_Wr = (const float*)d_in[5];
  const float* c1r_Wl = (const float*)d_in[6];
  const float* c1r_bl = (const float*)d_in[7];
  const float* c1r_Wr = (const float*)d_in[8];
  const float* ln_g   = (const float*)d_in[9];
  const float* ln_b   = (const float*)d_in[10];
  const float* c2f_Wl = (const float*)d_in[11];
  const float* c2f_bl = (const float*)d_in[12];
  const float* c2f_Wr = (const float*)d_in[13];
  const float* c2r_Wl = (const float*)d_in[14];
  const float* c2r_bl = (const float*)d_in[15];
  const float* c2r_Wr = (const float*)d_in[16];
  const float* d_W1   = (const float*)d_in[17];
  const float* d_b1   = (const float*)d_in[18];
  const float* d_W2   = (const float*)d_in[19];
  const float* d_b2   = (const float*)d_in[20];
  const float* d_W3   = (const float*)d_in[21];
  const float* d_b3   = (const float*)d_in[22];
  float* out = (float*)d_out;

  // Workspace carve-up (256B aligned)
  char* ws = (char*)d_ws;
  size_t off = 0;
  auto carve = [&](size_t bytes) { char* p = ws + off; off += (bytes + 255) & ~(size_t)255; return p; };
  float*    deg_f  = (float*)carve(N_NODES * 4);
  float*    deg_r  = (float*)carve(N_NODES * 4);
  float*    sum_f  = (float*)carve((size_t)N_NODES * D_IN * 4);
  float*    sum_r  = (float*)carve((size_t)N_NODES * D_IN * 4);
  _Float16* A1     = (_Float16*)carve((size_t)NPAD * A1_COLS * 2);
  float*    h      = (float*)carve((size_t)N_NODES * HID * 4);
  float*    sumh_f = (float*)carve((size_t)N_NODES * HID * 4);
  float*    sumh_r = (float*)carve((size_t)N_NODES * HID * 4);
  _Float16* A2     = (_Float16*)carve((size_t)NPAD * A2_COLS * 2);
  float*    z      = (float*)carve((size_t)N_NODES * OUT_D * 4);
  _Float16* comb   = (_Float16*)carve((size_t)N_LABEL * COMB_COLS * 2);
  _Float16* W1p    = (_Float16*)carve(192 * 128 * 2);
  float*    bias1  = (float*)carve(128 * 4);
  _Float16* W2p    = (_Float16*)carve(384 * 64 * 2);
  float*    bias2  = (float*)carve(64 * 4);
  _Float16* dW1h   = (_Float16*)carve(192 * 64 * 2);
  _Float16* dW2h   = (_Float16*)carve(64 * 32 * 2);

  // Zero accumulators every call (graph replays must be deterministic).
  hipMemsetAsync(deg_f, 0, N_NODES * 4, stream);
  hipMemsetAsync(deg_r, 0, N_NODES * 4, stream);
  hipMemsetAsync(sum_f, 0, (size_t)N_NODES * D_IN * 4, stream);
  hipMemsetAsync(sum_r, 0, (size_t)N_NODES * D_IN * 4, stream);
  hipMemsetAsync(sumh_f, 0, (size_t)N_NODES * HID * 4, stream);
  hipMemsetAsync(sumh_r, 0, (size_t)N_NODES * HID * 4, stream);

  // Weight packing (f32 -> f16, relation folding, WMMA fragment swizzle)
  pack_w1_kernel<<<(192 * 128 + 255) / 256, 256, 0, stream>>>(
      c1f_Wl, c1f_Wr, c1r_Wl, c1r_Wr, c1f_bl, c1r_bl, W1p, bias1);
  pack_w2_kernel<<<(384 * 64 + 255) / 256, 256, 0, stream>>>(
      c2f_Wl, c2f_Wr, c2r_Wl, c2r_Wr, c2f_bl, c2r_bl, W2p, bias2);
  pack_dec_kernel<<<(192 * 64 + 64 * 32 + 255) / 256, 256, 0, stream>>>(
      d_W1, d_W2, dW1h, dW2h);

  // Degrees + layer-1 edge scatter (mean aggregation numerators)
  deg_kernel<<<(N_EDGES + 255) / 256, 256, 0, stream>>>(ei, deg_f, deg_r);
  scatter_kernel<D_IN><<<(N_EDGES * (D_IN / 4) + 255) / 256, 256, 0, stream>>>(
      ei, x, sum_f, sum_r);

  // Layer 1: pack A, fused GEMM + LayerNorm + ReLU
  pack_a_kernel<D_IN><<<(NPAD * A1_COLS + 255) / 256, 256, 0, stream>>>(
      x, sum_f, deg_f, sum_r, deg_r, A1);
  gemm1_ln_relu_kernel<<<(NPAD / 16 + 3) / 4, 128, 0, stream>>>(
      A1, W1p, bias1, ln_g, ln_b, h);

  // Layer 2: scatter h, pack A, fused GEMM + L2 normalize
  scatter_kernel<HID><<<(N_EDGES * (HID / 4) + 255) / 256, 256, 0, stream>>>(
      ei, h, sumh_f, sumh_r);
  pack_a_kernel<HID><<<(NPAD * A2_COLS + 255) / 256, 256, 0, stream>>>(
      h, sumh_f, deg_f, sumh_r, deg_r, A2);
  gemm2_l2norm_kernel<<<(NPAD / 16 + 3) / 4, 128, 0, stream>>>(A2, W2p, bias2, z);

  // Decoder: gather label-edge embeddings, fused 3-layer MLP
  pack_comb_kernel<<<(N_LABEL * COMB_COLS + 255) / 256, 256, 0, stream>>>(eli, z, comb);
  decoder_kernel<<<(N_LABEL / 16 + 3) / 4, 128, 0, stream>>>(
      comb, dW1h, d_b1, dW2h, d_b2, d_W3, d_b3, out);
}
